// GraphCNN_28183575396967
// MI455X (gfx1250) — matrix operand order repaired and checked by reading, not compiled, not run on hardware
//
#include <hip/hip_runtime.h>
#include <hip/hip_bf16.h>

// ---------------------------------------------------------------------------
// GIN-style GraphCNN forward for MI455X (gfx1250, wave32).
// Heavy math via V_WMMA_F32_16X16X32_BF16 (f32 accumulate).
// A (adj, 256 MB) is streamed f32 and converted to bf16 in-flight (RNE,
// packed with V_PERM_B32); B (small, reused by every M-block) is converted
// AND transposed to bf16 [Nc][K] once per GEMM by a tiny prep kernel, so the
// GEMM's B staging is pure b128 data movement with zero conversion VALU.
// Double-buffered LDS overlaps global loads with WMMA; one barrier per iter.
// All M dims (8192, 64) are exact multiples of BM=64 -> no row guards.
// ---------------------------------------------------------------------------

typedef __attribute__((ext_vector_type(16))) __bf16 v16bf;
typedef __attribute__((ext_vector_type(8)))  float  v8f;

#define NNODES 8192
#define HIDD   256

// Round-to-nearest-even f32 -> bf16 for a pair of floats, packed into u32
// (x -> low 16, y -> high 16). ~5 VALU: 2x bfe, 2x add3, 1x v_perm_b32.
__device__ __forceinline__ unsigned int pack2_bf16_rne(float x, float y) {
  unsigned int ux = __float_as_uint(x);
  unsigned int uy = __float_as_uint(y);
  ux += 0x7FFFu + ((ux >> 16) & 1u);
  uy += 0x7FFFu + ((uy >> 16) & 1u);
  return __builtin_amdgcn_perm(uy, ux, 0x07060302u);
}

__device__ __forceinline__ unsigned short f2bf(float f) {
  unsigned int u = __float_as_uint(f);
  return (unsigned short)((u + 0x7FFFu + ((u >> 16) & 1u)) >> 16);
}

// ---------------------------------------------------------------------------
// B prep: f32 [K][Nc] row-major -> bf16 [Nc][K] (transposed), RNE.
// 32x32 LDS tiles; coalesced loads and stores. K%32==0, Nc%32==0.
// ---------------------------------------------------------------------------
__global__ __launch_bounds__(256) void gin_transpose_bf16(
    const float* __restrict__ B, unsigned short* __restrict__ Bt, int K, int Nc)
{
  __shared__ unsigned short tile[32][33];   // padded against bank conflicts
  const int k0  = blockIdx.x * 32;
  const int n0  = blockIdx.y * 32;
  const int col = threadIdx.x & 31;
  const int r8  = threadIdx.x >> 5;         // 0..7
#pragma unroll
  for (int i = 0; i < 4; ++i) {
    int r = r8 + 8 * i;                     // k-local
    tile[r][col] = f2bf(B[(size_t)(k0 + r) * Nc + (n0 + col)]);
  }
  __syncthreads();
#pragma unroll
  for (int i = 0; i < 4; ++i) {
    int r = r8 + 8 * i;                     // n-local
    Bt[(size_t)(n0 + r) * K + (k0 + col)] = tile[col][r];
  }
}

// ---------------------------------------------------------------------------
// C[M x Nc] = A[M x K] @ B[K x Nc] (+ bias); A f32, B pre-transposed bf16.
// Block tile: 64(M) x 128(N), BK = 32, 256 threads = 8 waves (2M x 4N),
// each wave owns a 32x32 region = 2x2 WMMA tiles.
// Requires: M % 64 == 0, Nc % 128 == 0, K % 32 == 0.
// ---------------------------------------------------------------------------
#define BM 64
#define BN 128
#define BK 32
#define LDA 40   // padded bf16 row stride (keeps 16B alignment of 8-elem chunks)
#define LDB 40

__global__ __launch_bounds__(256) void gin_gemm_bf16(
    const float* __restrict__ A, const unsigned short* __restrict__ Bt,
    const float* __restrict__ bias, float* __restrict__ C,
    int M, int K, int Nc)
{
  __shared__ unsigned short lA [2][BM * LDA];   // [row][k]
  __shared__ unsigned short lBt[2][BN * LDB];   // [n][k]

  const int tid  = threadIdx.x;
  const int lane = tid & 31;
  const int wave = tid >> 5;
  const int half = lane >> 4;   // lane group per ISA WMMA layout
  const int l16  = lane & 15;
  const int wm   = wave & 1;    // 2 waves along M (32 rows each)
  const int wn   = wave >> 1;   // 4 waves along N (32 cols each)

  const int bM = blockIdx.y * BM;
  const int bN = blockIdx.x * BN;

  v8f acc[2][2] = {};

  // staging registers for the in-flight tile
  float4 aReg[2];
  uint4  bReg[2];
  const int bn_row = tid >> 1;          // 0..127 (n within tile)
  const int bk_hr  = (tid & 1) << 4;    // 0 or 16 (k half-chunk)

  auto loadTiles = [&](int k0) {
#pragma unroll
    for (int i = 0; i < 2; ++i) {
      int idx = tid + 256 * i;               // 0..511
      int r   = idx >> 3;                    // 0..63
      int kq  = (idx & 7) << 2;              // 0,4,...,28
      aReg[i] = *(const float4*)(A + (size_t)(bM + r) * K + (k0 + kq));
    }
    const uint4* pb =
        (const uint4*)(Bt + (size_t)(bN + bn_row) * K + (k0 + bk_hr));
    bReg[0] = pb[0];                         // 16 bf16 = two b128 loads
    bReg[1] = pb[1];
  };
  auto storeTiles = [&](unsigned short* sA, unsigned short* sBt) {
#pragma unroll
    for (int i = 0; i < 2; ++i) {
      int idx = tid + 256 * i;
      int r   = idx >> 3;
      int kq  = (idx & 7) << 2;
      unsigned int a0 = pack2_bf16_rne(aReg[i].x, aReg[i].y);
      unsigned int a1 = pack2_bf16_rne(aReg[i].z, aReg[i].w);
      *(uint2*)&sA[r * LDA + kq] = make_uint2(a0, a1);   // ds_store_b64
    }
    uint4* d = (uint4*)&sBt[bn_row * LDB + bk_hr];       // 16B aligned
    d[0] = bReg[0];                                      // ds_store_b128 x2
    d[1] = bReg[1];
  };

  loadTiles(0);
  storeTiles(lA[0], lBt[0]);
  __syncthreads();

  int cur = 0;
  for (int k0 = 0; k0 < K; k0 += BK) {
    const bool more = (k0 + BK) < K;
    if (more) loadTiles(k0 + BK);            // HBM latency hides behind WMMA

    union Frag { v16bf v; unsigned short s[16]; };
    // A fragment (ISA 7.12.2 16-bit A layout): elems 0..7 -> K = e + 8*half,
    // elems 8..15 -> K = e + 16 + 8*half  (two contiguous 16B chunks).
    Frag af[2];
#pragma unroll
    for (int mt = 0; mt < 2; ++mt) {
      const unsigned short* pa = &lA[cur][(wm * 32 + mt * 16 + l16) * LDA];
#pragma unroll
      for (int e = 0; e < 8; ++e)  af[mt].s[e]     = pa[e + 8 * half];
#pragma unroll
      for (int e = 0; e < 8; ++e)  af[mt].s[e + 8] = pa[e + 16 + 8 * half];
    }
#pragma unroll
    for (int nt = 0; nt < 2; ++nt) {
      // B fragment: lane = col n, elems 0..7 -> K = e + 8*half, elems 8..15 -> K = e + 8 + 8*half.
      Frag bfr;
      const unsigned short* pb = &lBt[cur][(wn * 32 + nt * 16 + l16) * LDB];
#pragma unroll
      for (int e = 0; e < 16; ++e) bfr.s[e] = pb[(e < 8 ? e : e + 8) + 8 * half];
#pragma unroll
      for (int mt = 0; mt < 2; ++mt) {
        acc[mt][nt] = __builtin_amdgcn_wmma_f32_16x16x32_bf16(
            false, af[mt].v, false, bfr.v, (short)0, acc[mt][nt], false, false);
      }
    }

    if (more) storeTiles(lA[cur ^ 1], lBt[cur ^ 1]);
    __syncthreads();
    cur ^= 1;
  }

  // ---- epilogue: C/D layout — VGPR r holds row r + 8*half, col = lane%16 ----
#pragma unroll
  for (int mt = 0; mt < 2; ++mt) {
#pragma unroll
    for (int nt = 0; nt < 2; ++nt) {
      int col = bN + wn * 32 + nt * 16 + l16;
      float bv = bias ? bias[col] : 0.f;
      int rbase = bM + wm * 32 + mt * 16 + 8 * half;
#pragma unroll
      for (int r = 0; r < 8; ++r) {
        C[(size_t)(rbase + r) * Nc + col] = acc[mt][nt][r] + bv;
      }
    }
  }
}

// ---------------------------------------------------------------------------
// Deterministic two-stage BatchNorm stats (no float atomics -> bit-stable
// across graph replays). z is [NNODES x HIDD].
// ---------------------------------------------------------------------------
__global__ __launch_bounds__(256) void gin_col_partial(
    const float* __restrict__ z, float* __restrict__ psum, float* __restrict__ psq)
{
  int c = threadIdx.x;                  // column 0..255 (coalesced)
  int b = blockIdx.x;                   // 256 blocks x 32 rows
  const float* p = z + (size_t)b * 32 * HIDD + c;
  float s = 0.f, ss = 0.f;
#pragma unroll 4
  for (int r = 0; r < 32; ++r) { float v = p[(size_t)r * HIDD]; s += v; ss += v * v; }
  psum[b * HIDD + c] = s;
  psq [b * HIDD + c] = ss;
}

__global__ void gin_finalize_stats(
    const float* __restrict__ psum, const float* __restrict__ psq,
    const float* __restrict__ g, const float* __restrict__ be,
    float* __restrict__ scale, float* __restrict__ shift)
{
  int c = threadIdx.x;
  float s = 0.f, ss = 0.f;
  for (int b = 0; b < 256; ++b) { s += psum[b * HIDD + c]; ss += psq[b * HIDD + c]; }
  const float invM = 1.0f / (float)NNODES;
  float mean = s * invM;
  float var  = ss * invM - mean * mean;   // biased variance (training-mode BN)
  float sc   = g[c] * rsqrtf(var + 1e-5f);
  scale[c] = sc;
  shift[c] = be[c] - mean * sc;
}

__global__ __launch_bounds__(256) void gin_bn_relu(
    const float* __restrict__ z, const float* __restrict__ scale,
    const float* __restrict__ shift, float* __restrict__ y)
{
  int c = threadIdx.x;
  size_t i = (size_t)blockIdx.x * HIDD + c;
  y[i] = fmaxf(0.f, z[i] * scale[c] + shift[c]);
}

// ---------------------------------------------------------------------------
// Driver
// ---------------------------------------------------------------------------
extern "C" void kernel_launch(void* const* d_in, const int* in_sizes, int n_in,
                              void* d_out, int out_size, void* d_ws, size_t ws_size,
                              hipStream_t stream)
{
  (void)in_sizes; (void)n_in; (void)out_size; (void)ws_size;

  const float* x   = (const float*)d_in[0];   // [8192 x 128]
  const float* adj = (const float*)d_in[1];   // [8192 x 8192]
  const float* gp  = (const float*)d_in[2];   // [64 x 8192]
  const float* prm[16];
  for (int i = 0; i < 16; ++i) prm[i] = (const float*)d_in[3 + i];
  // per layer l: prm[8l+0]=W1, +1=b1, +2=g1, +3=be1, +4=W2, +5=b2, +6=g2, +7=be2

  float* ws = (float*)d_ws;                       // needs ~40.5 MB
  float* pooled = ws;                             // 8192*256 f32
  float* zbuf   = ws + (size_t)2 * 1024 * 1024;   // 8192*256
  float* tbuf   = ws + (size_t)4 * 1024 * 1024;   // 8192*256
  float* hbuf   = ws + (size_t)6 * 1024 * 1024;   // 8192*256
  float* psum   = ws + (size_t)8 * 1024 * 1024;   // 256*256
  float* psq    = psum + 256 * 256;
  float* scale  = psq + 256 * 256;
  float* shift  = scale + 256;
  unsigned short* bt = (unsigned short*)(ws + (size_t)9 * 1024 * 1024); // <=4MB

  float* out      = (float*)d_out;
  float* pooled_h = out;              // [64 x 256]   (first tuple element)
  float* h_nodes  = out + 64 * 256;   // [8192 x 256] (second tuple element)

  auto gemm = [&](const float* A, const float* B, const float* bias, float* C,
                  int M, int K, int Nc) {
    dim3 tg(K / 32, Nc / 32);
    gin_transpose_bf16<<<tg, 256, 0, stream>>>(B, bt, K, Nc);
    dim3 grid(Nc / BN, M / BM);
    gin_gemm_bf16<<<grid, 256, 0, stream>>>(A, bt, bias, C, M, K, Nc);
  };
  auto bnrelu = [&](const float* zz, const float* g, const float* be, float* y) {
    gin_col_partial<<<256, 256, 0, stream>>>(zz, psum, psq);
    gin_finalize_stats<<<1, 256, 0, stream>>>(psum, psq, g, be, scale, shift);
    gin_bn_relu<<<NNODES, 256, 0, stream>>>(zz, scale, shift, y);
  };

  // -------- layer 0 --------
  gemm(adj, x, nullptr, pooled, NNODES, NNODES, 128);       // neighbor sum
  gemm(pooled, prm[0], prm[1], zbuf, NNODES, 128, 256);     // @W1 + b1
  bnrelu(zbuf, prm[2], prm[3], tbuf);                       // inner BN + ReLU
  gemm(tbuf, prm[4], prm[5], zbuf, NNODES, 256, 256);       // @W2 + b2
  bnrelu(zbuf, prm[6], prm[7], hbuf);                       // outer BN + ReLU

  // -------- layer 1 --------
  gemm(adj, hbuf, nullptr, pooled, NNODES, NNODES, 256);
  gemm(pooled, prm[8], prm[9], zbuf, NNODES, 256, 256);
  bnrelu(zbuf, prm[10], prm[11], tbuf);
  gemm(tbuf, prm[12], prm[13], zbuf, NNODES, 256, 256);
  bnrelu(zbuf, prm[14], prm[15], h_nodes);                  // h_nodes -> d_out

  // -------- graph readout (M = 64 = BM exactly, no guards needed) --------
  gemm(gp, h_nodes, nullptr, pooled_h, 64, NNODES, 256);
}
